// QLSTMGen_65481071398055
// MI455X (gfx1250) — compile-verified, matrix operand order checked by reading, not run on hardware
//
#include <hip/hip_runtime.h>

// ---------------- problem dims ----------------
#define T_STEPS 512
#define BATCH   64
#define INDIM   512
#define HID     1024
#define KTOT    (INDIM + HID)          // 1536
#define NGATE   4
#define ZCOLS   (NGATE * HID)          // 4096

// ---------------- tiling ----------------
#define NWG     64                     // persistent workgroups (1 per WGP)
#define SLICE   (HID / NWG)            // 16 hidden units per WG
#define NCOLS   (NGATE * SLICE)        // 64 z-columns per WG
#define WKL     1544                   // LDS K-stride (1536 + 8 pad) in elements
#define ZLD     68                     // LDS z row stride (64 + 4 pad) in floats
#define THREADS 256                    // 8 wave32

typedef __bf16 bf16;
typedef bf16  v16bf __attribute__((ext_vector_type(16)));
typedef bf16  v8bf  __attribute__((ext_vector_type(8)));
typedef float v8f   __attribute__((ext_vector_type(8)));

#define WMMA_BF16(A, B, C) \
  __builtin_amdgcn_wmma_f32_16x16x32_bf16(false, (A), false, (B), (short)0, (C), false, false)

__device__ __forceinline__ unsigned short f2bf(float f) {
  unsigned u = __float_as_uint(f);
  u += 0x7FFFu + ((u >> 16) & 1u);     // round-to-nearest-even
  return (unsigned short)(u >> 16);
}

// 32B fragment load via two 16B halves (LDS columns are 16B aligned, stride 1544 elem)
__device__ __forceinline__ v16bf ld16bf(const unsigned short* p) {
  union { v16bf v; v8bf h[2]; } u;
  u.h[0] = *(const v8bf*)(p);
  u.h[1] = *(const v8bf*)(p + 8);
  return u.v;
}

__device__ __forceinline__ float fast_sigmoid(float x) {
  return 1.0f / (1.0f + __expf(-x));
}
__device__ __forceinline__ float fast_tanh(float x) {
  float e = __expf(-2.0f * x);
  return (1.0f - e) / (1.0f + e);
}

// ---------------- prep kernels (run every launch; deterministic) ----------------
__global__ void init_state_kernel(unsigned short* hbuf, unsigned* bar) {
  int i = blockIdx.x * THREADS + threadIdx.x;
  if (i < 2 * BATCH * HID) hbuf[i] = 0;
  if (i < 4) bar[i] = 0u;
}

__global__ void x_to_bf16_kernel(const float* __restrict__ x,
                                 unsigned short* __restrict__ xb, int n) {
  int i = blockIdx.x * THREADS + threadIdx.x;
  if (i < n) xb[i] = f2bf(x[i]);
}

// Pack W = [Wf|Wi|Wg|Wo] ([1536,4096]) as bf16, column-major (K contiguous per column)
__global__ void w_pack_kernel(const float* __restrict__ Wf, const float* __restrict__ Wi,
                              const float* __restrict__ Wg, const float* __restrict__ Wo,
                              unsigned short* __restrict__ Wcat) {
  int i = blockIdx.x * THREADS + threadIdx.x;
  if (i >= ZCOLS * KTOT) return;
  int c = i & (ZCOLS - 1);             // global z column (coalesced reads over u)
  int k = i >> 12;
  int gate = c >> 10;
  int u = c & (HID - 1);
  const float* W = (gate == 0) ? Wf : (gate == 1) ? Wi : (gate == 2) ? Wg : Wo;
  Wcat[(size_t)c * KTOT + k] = f2bf(W[(size_t)k * HID + u]);
}

// ---------------- persistent LSTM kernel ----------------
__global__ __launch_bounds__(THREADS, 1)
void lstm_persistent_kernel(const unsigned short* __restrict__ Wcat,  // [4096][1536] col-major
                            const unsigned short* __restrict__ Xb,    // [T][B][512] bf16
                            const float* __restrict__ bfv, const float* __restrict__ biv,
                            const float* __restrict__ bgv, const float* __restrict__ bov,
                            unsigned short* __restrict__ hbuf,        // [2][B][HID] bf16 ping-pong
                            float* __restrict__ out,                  // stacked + hx + cx
                            unsigned* __restrict__ bar) {
  extern __shared__ char smem[];
  unsigned short* Wl = (unsigned short*)smem;                 // NCOLS * WKL bf16 (weights, stationary)
  float* Zl = (float*)(smem + (size_t)NCOLS * WKL * 2);       // BATCH * ZLD     (z tile)
  float* Cl = Zl + BATCH * ZLD;                               // BATCH * SLICE   (cell state, persistent)
  float* Bl = Cl + BATCH * SLICE;                             // NCOLS           (bias)

  const int wg    = blockIdx.x;
  const int u0    = wg * SLICE;
  const int tid   = threadIdx.x;
  const int wave  = tid >> 5;
  const int lane  = tid & 31;
  const int laneM = lane & 15;
  const int kHi   = (lane >> 4) << 4;  // 0 / 16 : K half, contiguous k = 16h + s order (same for A and B)

  // ---- stage this WG's 64 weight columns into LDS (once for all 512 steps) ----
  for (int idx = tid; idx < NCOLS * (KTOT / 16); idx += THREADS) {
    int lc  = idx / (KTOT / 16);
    int k16 = (idx - lc * (KTOT / 16)) << 4;
    int c   = ((lc >> 4) << 10) + u0 + (lc & 15);   // gate*1024 + hidden unit
    const uint4* src = (const uint4*)(Wcat + (size_t)c * KTOT + k16);
    uint4* dst = (uint4*)(Wl + (size_t)lc * WKL + k16);
    dst[0] = src[0];
    dst[1] = src[1];
  }
  if (tid < NCOLS) {
    int g = tid >> 4;
    int u = u0 + (tid & 15);
    const float* bp = (g == 0) ? bfv : (g == 1) ? biv : (g == 2) ? bgv : bov;
    Bl[tid] = bp[u];
  }
  for (int i = tid; i < BATCH * SLICE; i += THREADS) Cl[i] = 0.0f;
  __syncthreads();

  // wave -> (row tile, 2 column tiles)
  const int mt   = wave >> 1;             // 0..3  (16 batch rows each)
  const int nt0  = (wave & 1) << 1;       // 0 or 2 (two 16-col tiles)
  const int rowA = (mt << 4) + laneM;

  const unsigned short* bc0 = Wl + (size_t)((nt0 << 4) + laneM) * WKL + kHi;
  const unsigned short* bc1 = Wl + (size_t)(((nt0 + 1) << 4) + laneM) * WKL + kHi;

  for (int t = 0; t < T_STEPS; ++t) {
    const unsigned short* xrow = Xb + ((size_t)t * BATCH + rowA) * INDIM + kHi;
    const unsigned short* hrow = hbuf + (size_t)(t & 1) * BATCH * HID + (size_t)rowA * HID + kHi;

    v8f acc0 = {0.f, 0.f, 0.f, 0.f, 0.f, 0.f, 0.f, 0.f};
    v8f acc1 = acc0;

    // z = x_t @ Wx  (K = 0..511)
#pragma unroll 4
    for (int kb = 0; kb < INDIM; kb += 32) {
      v16bf a  = ld16bf(xrow + kb);
      v16bf b0 = ld16bf(bc0 + kb);
      v16bf b1 = ld16bf(bc1 + kb);
      acc0 = WMMA_BF16(a, b0, acc0);
      acc1 = WMMA_BF16(a, b1, acc1);
    }
    // z += h_{t-1} @ Wh  (K = 512..1535)
#pragma unroll 4
    for (int kb = 0; kb < HID; kb += 32) {
      v16bf a  = ld16bf(hrow + kb);
      v16bf b0 = ld16bf(bc0 + INDIM + kb);
      v16bf b1 = ld16bf(bc1 + INDIM + kb);
      acc0 = WMMA_BF16(a, b0, acc0);
      acc1 = WMMA_BF16(a, b1, acc1);
    }

    __syncthreads();
    {
      // C/D layout: vgpr r holds M=r (lanes 0-15) / M=8+r (lanes 16-31), N=lane&15
      const int zr = (mt << 4) + ((lane >> 4) << 3);
      const int zc = (nt0 << 4) + laneM;
#pragma unroll
      for (int r = 0; r < 8; ++r) {
        Zl[(zr + r) * ZLD + zc]      = acc0[r];
        Zl[(zr + r) * ZLD + zc + 16] = acc1[r];
      }
    }
    __syncthreads();

    // ---- gates + state update (c stays in LDS; h -> global ping-pong + output) ----
    unsigned short* hnext = hbuf + (size_t)((t + 1) & 1) * BATCH * HID;
    const bool last = (t == T_STEPS - 1);
#pragma unroll
    for (int i = tid; i < BATCH * SLICE; i += THREADS) {
      int b = i >> 4;
      int u = i & 15;
      float zf = Zl[b * ZLD +  0 + u] + Bl[ 0 + u];
      float zi = Zl[b * ZLD + 16 + u] + Bl[16 + u];
      float zg = Zl[b * ZLD + 32 + u] + Bl[32 + u];
      float zo = Zl[b * ZLD + 48 + u] + Bl[48 + u];
      float fg = fast_sigmoid(zf);
      float ig = fast_sigmoid(zi);
      float gg = fast_tanh(zg);
      float og = fast_sigmoid(zo);
      float cn = fg * Cl[i] + ig * gg;
      Cl[i] = cn;
      float h = og * fast_tanh(cn);
      int hu = u0 + u;
      out[((size_t)t * BATCH + b) * HID + hu] = h;
      hnext[(size_t)b * HID + hu] = f2bf(h);
      if (last) {
        out[(size_t)T_STEPS * BATCH * HID + (size_t)b * HID + hu] = h;                        // hx
        out[(size_t)T_STEPS * BATCH * HID + (size_t)BATCH * HID + (size_t)b * HID + hu] = cn; // cx
      }
    }

    // prefetch next timestep's x tile (64KB) into cache while we wait at the barrier
    if (t + 1 < T_STEPS) {
      const char* xn = (const char*)(Xb + (size_t)(t + 1) * BATCH * INDIM);
      __builtin_prefetch(xn + tid * 256, 0, 0);
      __builtin_prefetch(xn + tid * 256 + 128, 0, 0);
    }

    // ---- device-wide sense-reversing barrier (all 64 WGs co-resident) ----
    __threadfence();
    __syncthreads();
    if (tid == 0) {
      volatile unsigned* gen = bar + 1;
      unsigned g = *gen;
      if (atomicAdd(bar, 1u) == NWG - 1) {
        atomicExch(bar, 0u);
        __threadfence();
        atomicAdd((unsigned*)(bar + 1), 1u);
      } else {
        while (*gen == g) { __builtin_amdgcn_s_sleep(2); }
      }
    }
    __syncthreads();
  }
}

// ---------------- host launch ----------------
extern "C" void kernel_launch(void* const* d_in, const int* in_sizes, int n_in,
                              void* d_out, int out_size, void* d_ws, size_t ws_size,
                              hipStream_t stream) {
  (void)in_sizes; (void)n_in; (void)out_size;
  const float* X  = (const float*)d_in[0];
  const float* Wf = (const float*)d_in[1];
  const float* bf = (const float*)d_in[2];
  const float* Wi = (const float*)d_in[3];
  const float* bi = (const float*)d_in[4];
  const float* Wg = (const float*)d_in[5];
  const float* bg = (const float*)d_in[6];
  const float* Wo = (const float*)d_in[7];
  const float* bo = (const float*)d_in[8];
  float* out = (float*)d_out;

  const size_t wcat_off = 0;
  const size_t wcat_sz  = (size_t)ZCOLS * KTOT * 2;                 // 12.6 MB bf16 weights
  const size_t xb_off   = wcat_off + wcat_sz;
  const size_t xb_sz    = (size_t)T_STEPS * BATCH * INDIM * 2;      // 33.6 MB bf16 inputs
  const size_t h_off    = xb_off + xb_sz;
  const size_t h_sz     = (size_t)2 * BATCH * HID * 2;              // h ping-pong
  const size_t bar_off  = h_off + h_sz;
  const size_t total    = bar_off + 256;
  if (ws_size < total) return;

  unsigned short* Wcat = (unsigned short*)((char*)d_ws + wcat_off);
  unsigned short* Xb   = (unsigned short*)((char*)d_ws + xb_off);
  unsigned short* hbuf = (unsigned short*)((char*)d_ws + h_off);
  unsigned*       bar  = (unsigned*)((char*)d_ws + bar_off);

  init_state_kernel<<<(2 * BATCH * HID + THREADS - 1) / THREADS, THREADS, 0, stream>>>(hbuf, bar);
  x_to_bf16_kernel<<<(T_STEPS * BATCH * INDIM + THREADS - 1) / THREADS, THREADS, 0, stream>>>(
      X, Xb, T_STEPS * BATCH * INDIM);
  w_pack_kernel<<<(ZCOLS * KTOT + THREADS - 1) / THREADS, THREADS, 0, stream>>>(Wf, Wi, Wg, Wo, Wcat);

  const size_t smem = (size_t)NCOLS * WKL * 2      // 197,632 B weights
                    + (size_t)BATCH * ZLD * 4      //  17,408 B z tile
                    + (size_t)BATCH * SLICE * 4    //   4,096 B cell state
                    + (size_t)NCOLS * 4;           //     256 B bias  => 219,392 B < 320 KB
  (void)hipFuncSetAttribute(reinterpret_cast<const void*>(lstm_persistent_kernel),
                            hipFuncAttributeMaxDynamicSharedMemorySize, (int)smem);
  lstm_persistent_kernel<<<NWG, THREADS, smem, stream>>>(Wcat, Xb, bf, bi, bg, bo, hbuf, out, bar);
}
